// DeepDrugNet_V4_25142738551010
// MI455X (gfx1250) — compile-verified, matrix operand order checked by reading, not run on hardware
//
#include <hip/hip_runtime.h>
#include <hip/hip_bf16.h>

typedef _Float16 v16h __attribute__((ext_vector_type(16)));
typedef _Float16 v8h  __attribute__((ext_vector_type(8)));
typedef float    v8f  __attribute__((ext_vector_type(8)));

#define N_NODES 65536
#define N_EDGES 1048576
#define BATCH_B 512
#define PROT_L  512
#define EPS_BN  1e-5f

#define GMODE_F32 0
#define GMODE_AFF 1
#define GMODE_MAX 2

__device__ __forceinline__ void atomicMaxF(float* addr, float v) {
    if (v >= 0.f) atomicMax((int*)addr, __float_as_int(v));
    else          atomicMin((unsigned int*)addr, __float_as_uint(v));
}

// ---------------- Weight prep kernels ----------------

// GAT1 tables: t1h[10,64] = emb @ g1W ; t1es/t1ed [10,2]
__global__ void k_prep_gat1(const float* __restrict__ emb, const float* __restrict__ W,
                            const float* __restrict__ as_, const float* __restrict__ ad_,
                            float* __restrict__ t1h, float* __restrict__ t1es, float* __restrict__ t1ed) {
    int j = threadIdx.x; // 0..63
    for (int v = 0; v < 10; ++v) {
        float s = 0.f;
        for (int k = 0; k < 32; ++k) s += emb[v*32 + k] * W[k*64 + j];
        t1h[v*64 + j] = s;
    }
    __syncthreads();
    if (j < 20) {
        int v = j >> 1, h = j & 1;
        float s = 0.f, d = 0.f;
        for (int c = 0; c < 32; ++c) {
            float x = t1h[v*64 + h*32 + c];
            s += x * as_[h*32 + c];
            d += x * ad_[h*32 + c];
        }
        t1es[v*2 + h] = s; t1ed[v*2 + h] = d;
    }
}

// dst[o,k] (f16) = src[k,o] (f32)   -- [K,O] -> [O,K]
__global__ void k_transpose_kf(const float* __restrict__ src, _Float16* __restrict__ dst, int K, int O) {
    int gid = blockIdx.x * blockDim.x + threadIdx.x;
    if (gid >= O * K) return;
    int o = gid / K, k = gid - o * K;
    dst[gid] = (_Float16)src[k * O + o];
}

// conv weight: src [O,C,3] row-major -> dst [O, 3C] with dst[o, k*C+i] = src[o,i,k]
__global__ void k_conv_w(const float* __restrict__ src, _Float16* __restrict__ dst, int O, int C) {
    int K = 3 * C;
    int gid = blockIdx.x * blockDim.x + threadIdx.x;
    if (gid >= O * K) return;
    int o = gid / K, r = gid - o * K;
    int k = r / C, i = r - k * C;
    dst[gid] = (_Float16)src[o * K + i * 3 + k];
}

// BN fold: scale = g*rsqrt(v+eps); shift = (bias - m)*scale + b   (bias may be null)
__global__ void k_affine(const float* __restrict__ g, const float* __restrict__ b,
                         const float* __restrict__ m, const float* __restrict__ v,
                         const float* __restrict__ bias,
                         float* __restrict__ scale, float* __restrict__ shift, int O) {
    int o = blockIdx.x * blockDim.x + threadIdx.x;
    if (o >= O) return;
    float sc = g[o] * rsqrtf(v[o] + EPS_BN);
    float bi = bias ? bias[o] : 0.f;
    scale[o] = sc;
    shift[o] = (bi - m[o]) * sc + b[o];
}

// ---------------- Generic WMMA GEMM ----------------
// C[m, n] = sum_k A[m, k] * Wt[n, k], A rows addressed via m -> (b = m/Lout, l = m%Lout),
// row offset (b*Lin + l)*Cin elements (implicit-GEMM conv patch; for plain GEMM Lout=M, Cin=K).
// K multiple of 32, M multiple of 16, O multiple of 16. Block = 32*(O/16) threads, one M-tile/block.
__global__ __launch_bounds__(256) void k_gemm_wmma(
    const _Float16* __restrict__ A, const _Float16* __restrict__ Wt,
    int K, int Lout, int Lin, int Cin, int O, int mode,
    float* __restrict__ outF32, _Float16* __restrict__ outF16,
    const float* __restrict__ scale, const float* __restrict__ shift,
    float* __restrict__ prot) {
    __shared__ __align__(16) _Float16 As[16 * 384];
    const int tid = threadIdx.x;
    const int mt = blockIdx.x;
    const int k8 = K >> 3;
    // Stage A tile (16 rows x K halves) into LDS, 16B chunks
    for (int ch = tid; ch < 16 * k8; ch += blockDim.x) {
        int r = ch / k8, c8 = ch - r * k8;
        int m = mt * 16 + r;
        int b = m / Lout, l = m - b * Lout;
        const uint4* src = (const uint4*)(A + (long long)(b * Lin + l) * Cin + c8 * 8);
        *(uint4*)(&As[r * K + c8 * 8]) = *src;
    }
    __syncthreads();

    const int lane = tid & 31;
    const int wv = tid >> 5;               // N-tile index (wave32)
    const int col = lane & 15;
    const int kb = (lane & 16) ? 8 : 0;    // per-ISA 16-bit A/B fragment layout
    const _Float16* Arow = As + (lane & 15) * K;
    const _Float16* Brow = Wt + (long long)(wv * 16 + col) * K;

    v8f c = {};
    for (int k0 = 0; k0 < K; k0 += 32) {
        v8h a0 = *(const v8h*)(Arow + k0 + kb);
        v8h a1 = *(const v8h*)(Arow + k0 + 16 + kb);
        v8h b0 = *(const v8h*)(Brow + k0 + kb);
        v8h b1 = *(const v8h*)(Brow + k0 + 16 + kb);
        v16h av, bv;
#pragma unroll
        for (int i = 0; i < 8; ++i) { av[i] = a0[i]; av[i+8] = a1[i]; bv[i] = b0[i]; bv[i+8] = b1[i]; }
        c = __builtin_amdgcn_wmma_f32_16x16x32_f16(false, av, false, bv, (short)0, c, false, false);
    }

    const int colg = wv * 16 + col;
    const int mbase = (lane & 16) ? 8 : 0;
#pragma unroll
    for (int i = 0; i < 8; ++i) {
        int m = mt * 16 + mbase + i;
        float v = c[i];
        if (mode == GMODE_F32) {
            outF32[(long long)m * O + colg] = v;
        } else {
            v = fmaxf(v * scale[colg] + shift[colg], 0.f);
            if (mode == GMODE_AFF) {
                outF16[(long long)m * O + colg] = (_Float16)v;
            } else {
                int b = m / Lout;
                atomicMax((unsigned int*)&prot[b * O + colg], __float_as_uint(v));
            }
        }
    }
}

// ---------------- GAT kernels ----------------

__global__ void k_gat1_gather(const int* __restrict__ xn,
                              const float* __restrict__ t1h, const float* __restrict__ t1es,
                              const float* __restrict__ t1ed,
                              float* __restrict__ H, float* __restrict__ ES, float* __restrict__ ED) {
    int gid = blockIdx.x * blockDim.x + threadIdx.x;
    if (gid >= N_NODES * 64) return;
    int n = gid >> 6, j = gid & 63;
    int v = xn[n];
    H[gid] = t1h[v * 64 + j];
    if (j < 2) { ES[n * 2 + j] = t1es[v * 2 + j]; ED[n * 2 + j] = t1ed[v * 2 + j]; }
}

__global__ void k_es_ed(const float* __restrict__ H, const float* __restrict__ as_,
                        const float* __restrict__ ad_, float* __restrict__ ES, float* __restrict__ ED,
                        int Hh, int C) {
    int gid = blockIdx.x * blockDim.x + threadIdx.x;
    if (gid >= N_NODES * Hh) return;
    int n = gid / Hh, h = gid - n * Hh;
    const float* row = H + (long long)n * Hh * C + h * C;
    float s = 0.f, d = 0.f;
    for (int c = 0; c < C; ++c) { float x = row[c]; s += x * as_[h * C + c]; d += x * ad_[h * C + c]; }
    ES[gid] = s; ED[gid] = d;
}

__global__ void k_edge_a(const int* __restrict__ ei, int Hh,
                         const float* __restrict__ ES, const float* __restrict__ ED,
                         float* __restrict__ ESCR, float* __restrict__ MX) {
    int gid = blockIdx.x * blockDim.x + threadIdx.x;
    int total = (N_EDGES + N_NODES) * Hh;
    if (gid >= total) return;
    int e = gid / Hh, h = gid - e * Hh;
    int s, d;
    if (e < N_EDGES) { s = ei[e]; d = ei[N_EDGES + e]; } else { s = d = e - N_EDGES; }
    float v = ES[s * Hh + h] + ED[d * Hh + h];
    v = v > 0.f ? v : 0.2f * v;                 // leaky_relu 0.2
    ESCR[gid] = v;
    atomicMaxF(&MX[d * Hh + h], v);
}

__global__ void k_edge_b1(const int* __restrict__ ei, int Hh,
                          float* __restrict__ ESCR, const float* __restrict__ MX,
                          float* __restrict__ DEN) {
    int gid = blockIdx.x * blockDim.x + threadIdx.x;
    int total = (N_EDGES + N_NODES) * Hh;
    if (gid >= total) return;
    int e = gid / Hh, h = gid - e * Hh;
    int d = (e < N_EDGES) ? ei[N_EDGES + e] : (e - N_EDGES);
    float w = __expf(ESCR[gid] - MX[d * Hh + h]);
    ESCR[gid] = w;
    atomicAdd(&DEN[d * Hh + h], w);
}

__global__ void k_edge_b2(const int* __restrict__ ei, int Hh, int hcShift, int cShift,
                          const float* __restrict__ ESCR, const float* __restrict__ H,
                          float* __restrict__ ACC) {
    int gid = blockIdx.x * blockDim.x + threadIdx.x;
    int HC = 1 << hcShift;
    int total = (N_EDGES + N_NODES) << hcShift;
    if (gid >= total) return;
    int e = gid >> hcShift, c = gid & (HC - 1);
    int h = c >> cShift;
    int s, d;
    if (e < N_EDGES) { s = ei[e]; d = ei[N_EDGES + e]; } else { s = d = e - N_EDGES; }
    float w = ESCR[e * Hh + h];
    atomicAdd(&ACC[((long long)d << hcShift) + c], H[((long long)s << hcShift) + c] * w);
}

__global__ void k_node_fin(const float* __restrict__ ACC, const float* __restrict__ DEN,
                           const float* __restrict__ bias, int Hh, int hcShift, int cShift,
                           _Float16* __restrict__ XH, int doPool,
                           const int* __restrict__ batch, float* __restrict__ POOL,
                           float* __restrict__ CNT) {
    int gid = blockIdx.x * blockDim.x + threadIdx.x;
    int HC = 1 << hcShift;
    if (gid >= (N_NODES << hcShift)) return;
    int n = gid >> hcShift, c = gid & (HC - 1);
    int h = c >> cShift;
    float v = ACC[gid] / (DEN[n * Hh + h] + 1e-16f) + bias[c];
    v = v > 0.f ? v : (__expf(v) - 1.f);        // elu
    if (doPool) {
        int g = batch[n];
        atomicAdd(&POOL[g * 128 + c], v);
        if (c == 0) atomicAdd(&CNT[g], 1.f);
    } else {
        XH[gid] = (_Float16)v;
    }
}

// ---------------- Pool / protein / head kernels ----------------

__global__ void k_pool_fin(const float* __restrict__ POOL, const float* __restrict__ CNT,
                           const float* __restrict__ dsc, const float* __restrict__ dsh,
                           _Float16* __restrict__ Z) {
    int gid = blockIdx.x * blockDim.x + threadIdx.x;
    if (gid >= BATCH_B * 128) return;
    int g = gid >> 7, c = gid & 127;
    float mean = POOL[gid] / fmaxf(CNT[g], 1.f);
    Z[g * 256 + c] = (_Float16)(mean * dsc[c] + dsh[c]);
}

__global__ void k_prot_embed(const int* __restrict__ prot, const float* __restrict__ embp,
                             _Float16* __restrict__ PE) {
    int gid = blockIdx.x * blockDim.x + threadIdx.x;
    if (gid >= BATCH_B * PROT_L * 32) return;
    int i = gid >> 5, c = gid & 31;
    PE[gid] = (_Float16)embp[prot[i] * 32 + c];
}

__global__ void k_prot_to_z(const float* __restrict__ PROT, _Float16* __restrict__ Z) {
    int gid = blockIdx.x * blockDim.x + threadIdx.x;
    if (gid >= BATCH_B * 128) return;
    int b = gid >> 7, c = gid & 127;
    Z[b * 256 + 128 + c] = (_Float16)PROT[gid];
}

__global__ void k_out_head(const _Float16* __restrict__ Z2, const float* __restrict__ oW,
                           const float* __restrict__ ob, float* __restrict__ out) {
    int b = blockIdx.x * blockDim.x + threadIdx.x;
    if (b >= BATCH_B) return;
    float acc = ob[0];
    for (int j = 0; j < 64; ++j) acc += (float)Z2[b * 64 + j] * oW[j];
    out[b] = acc;
}

// ---------------- Launch ----------------

static inline int blk(long long n, int t) { return (int)((n + t - 1) / t); }

extern "C" void kernel_launch(void* const* d_in, const int* in_sizes, int n_in,
                              void* d_out, int out_size, void* d_ws, size_t ws_size,
                              hipStream_t stream) {
    const int*   xn     = (const int*)d_in[0];
    const int*   ei     = (const int*)d_in[1];
    const int*   batch  = (const int*)d_in[2];
    const int*   protein= (const int*)d_in[3];
    const float* emb_drug = (const float*)d_in[4];
    const float* g1W = (const float*)d_in[5];
    const float* g1as= (const float*)d_in[6];
    const float* g1ad= (const float*)d_in[7];
    const float* g1b = (const float*)d_in[8];
    const float* g2W = (const float*)d_in[9];
    const float* g2as= (const float*)d_in[10];
    const float* g2ad= (const float*)d_in[11];
    const float* g2b = (const float*)d_in[12];
    const float* g3W = (const float*)d_in[13];
    const float* g3as= (const float*)d_in[14];
    const float* g3ad= (const float*)d_in[15];
    const float* g3b = (const float*)d_in[16];
    const float* bn3g=(const float*)d_in[17]; const float* bn3b=(const float*)d_in[18];
    const float* bn3m=(const float*)d_in[19]; const float* bn3v=(const float*)d_in[20];
    const float* emb_prot=(const float*)d_in[21];
    const float* pc1W=(const float*)d_in[22]; const float* pc1b=(const float*)d_in[23];
    const float* pbn1g=(const float*)d_in[24]; const float* pbn1b=(const float*)d_in[25];
    const float* pbn1m=(const float*)d_in[26]; const float* pbn1v=(const float*)d_in[27];
    const float* pc2W=(const float*)d_in[28]; const float* pc2b=(const float*)d_in[29];
    const float* pbn2g=(const float*)d_in[30]; const float* pbn2b=(const float*)d_in[31];
    const float* pbn2m=(const float*)d_in[32]; const float* pbn2v=(const float*)d_in[33];
    const float* pc3W=(const float*)d_in[34]; const float* pc3b=(const float*)d_in[35];
    const float* pbn3g=(const float*)d_in[36]; const float* pbn3b=(const float*)d_in[37];
    const float* pbn3m=(const float*)d_in[38]; const float* pbn3v=(const float*)d_in[39];
    const float* fc1W=(const float*)d_in[40]; const float* fc1b=(const float*)d_in[41];
    const float* fbn1g=(const float*)d_in[42]; const float* fbn1b=(const float*)d_in[43];
    const float* fbn1m=(const float*)d_in[44]; const float* fbn1v=(const float*)d_in[45];
    const float* fc2W=(const float*)d_in[46]; const float* fc2b=(const float*)d_in[47];
    const float* fbn2g=(const float*)d_in[48]; const float* fbn2b=(const float*)d_in[49];
    const float* fbn2m=(const float*)d_in[50]; const float* fbn2v=(const float*)d_in[51];
    const float* outW=(const float*)d_in[52]; const float* outb=(const float*)d_in[53];

    char* w = (char*)d_ws;
    size_t off = 0;
    auto alloc = [&](size_t bytes) -> void* {
        void* p = w + off;
        off += (bytes + 255) & ~(size_t)255;
        return p;
    };

    float*    T1H  = (float*)alloc(10*64*4);
    float*    T1ES = (float*)alloc(10*2*4);
    float*    T1ED = (float*)alloc(10*2*4);
    _Float16* W2T  = (_Float16*)alloc(128*64*2);
    _Float16* W3T  = (_Float16*)alloc(128*128*2);
    _Float16* WC1T = (_Float16*)alloc(64*96*2);
    _Float16* WC2T = (_Float16*)alloc(128*192*2);
    _Float16* WC3T = (_Float16*)alloc(128*384*2);
    _Float16* F1T  = (_Float16*)alloc(128*256*2);
    _Float16* F2T  = (_Float16*)alloc(64*128*2);
    float* SC1=(float*)alloc(64*4);  float* SH1=(float*)alloc(64*4);
    float* SC2=(float*)alloc(128*4); float* SH2=(float*)alloc(128*4);
    float* SC3=(float*)alloc(128*4); float* SH3=(float*)alloc(128*4);
    float* SCF1=(float*)alloc(128*4);float* SHF1=(float*)alloc(128*4);
    float* SCF2=(float*)alloc(64*4); float* SHF2=(float*)alloc(64*4);
    float* DSC=(float*)alloc(128*4); float* DSH=(float*)alloc(128*4);
    float* ES  = (float*)alloc((size_t)N_NODES*2*4);
    float* ED  = (float*)alloc((size_t)N_NODES*2*4);
    float* MX  = (float*)alloc((size_t)N_NODES*2*4);
    float* DEN = (float*)alloc((size_t)N_NODES*2*4);
    float* POOL= (float*)alloc((size_t)BATCH_B*128*4);
    float* CNT = (float*)alloc((size_t)BATCH_B*4);
    float* PROT= (float*)alloc((size_t)BATCH_B*128*4);
    _Float16* Z  = (_Float16*)alloc((size_t)BATCH_B*256*2);
    _Float16* Z1 = (_Float16*)alloc((size_t)BATCH_B*128*2);
    _Float16* Z2 = (_Float16*)alloc((size_t)BATCH_B*64*2);
    float* ESCR = (float*)alloc((size_t)(N_EDGES+N_NODES)*2*4);
    // Big region: GAT phase {XH,H,ACC} overlapped with protein phase {PE,Y1,Y2}
    char* big = (char*)alloc(116916224ULL);
    _Float16* XH  = (_Float16*)(big + 0);
    float*    Hb  = (float*)(big + 16777216);
    float*    ACC = (float*)(big + 50331648);
    _Float16* PE  = (_Float16*)(big + 0);
    _Float16* Y1  = (_Float16*)(big + 16777216);
    _Float16* Y2  = (_Float16*)(big + 50331648);

    const int T = 256;

    // ---- weight prep ----
    k_prep_gat1<<<1, 64, 0, stream>>>(emb_drug, g1W, g1as, g1ad, T1H, T1ES, T1ED);
    k_transpose_kf<<<blk(128*64,T),T,0,stream>>>(g2W, W2T, 64, 128);
    k_transpose_kf<<<blk(128*128,T),T,0,stream>>>(g3W, W3T, 128, 128);
    k_transpose_kf<<<blk(128*256,T),T,0,stream>>>(fc1W, F1T, 256, 128);
    k_transpose_kf<<<blk(64*128,T),T,0,stream>>>(fc2W, F2T, 128, 64);
    k_conv_w<<<blk(64*96,T),T,0,stream>>>(pc1W, WC1T, 64, 32);
    k_conv_w<<<blk(128*192,T),T,0,stream>>>(pc2W, WC2T, 128, 64);
    k_conv_w<<<blk(128*384,T),T,0,stream>>>(pc3W, WC3T, 128, 128);
    k_affine<<<1,64,0,stream>>>(pbn1g,pbn1b,pbn1m,pbn1v,pc1b,SC1,SH1,64);
    k_affine<<<1,128,0,stream>>>(pbn2g,pbn2b,pbn2m,pbn2v,pc2b,SC2,SH2,128);
    k_affine<<<1,128,0,stream>>>(pbn3g,pbn3b,pbn3m,pbn3v,pc3b,SC3,SH3,128);
    k_affine<<<1,128,0,stream>>>(fbn1g,fbn1b,fbn1m,fbn1v,fc1b,SCF1,SHF1,128);
    k_affine<<<1,64,0,stream>>>(fbn2g,fbn2b,fbn2m,fbn2v,fc2b,SCF2,SHF2,64);
    k_affine<<<1,128,0,stream>>>(bn3g,bn3b,bn3m,bn3v,nullptr,DSC,DSH,128);

    hipMemsetAsync(POOL, 0, (size_t)BATCH_B*128*4, stream);
    hipMemsetAsync(CNT,  0, (size_t)BATCH_B*4, stream);
    hipMemsetAsync(PROT, 0, (size_t)BATCH_B*128*4, stream);

    const long long ET = (long long)(N_EDGES + N_NODES);

    // ---- GAT layer 1 (H=2, C=32, HC=64) ----
    k_gat1_gather<<<blk((long long)N_NODES*64,T),T,0,stream>>>(xn, T1H, T1ES, T1ED, Hb, ES, ED);
    hipMemsetAsync(MX, 0xFF, (size_t)N_NODES*2*4, stream);
    hipMemsetAsync(DEN, 0,   (size_t)N_NODES*2*4, stream);
    hipMemsetAsync(ACC, 0,   (size_t)N_NODES*64*4, stream);
    k_edge_a <<<blk(ET*2,T),T,0,stream>>>(ei, 2, ES, ED, ESCR, MX);
    k_edge_b1<<<blk(ET*2,T),T,0,stream>>>(ei, 2, ESCR, MX, DEN);
    k_edge_b2<<<blk(ET*64,T),T,0,stream>>>(ei, 2, 6, 5, ESCR, Hb, ACC);
    k_node_fin<<<blk((long long)N_NODES*64,T),T,0,stream>>>(ACC, DEN, g1b, 2, 6, 5, XH, 0, batch, POOL, CNT);

    // ---- GAT layer 2 (H=2, C=64, HC=128): h = XH[N,64] @ W2T ----
    k_gemm_wmma<<<N_NODES/16, 256, 0, stream>>>(XH, W2T, 64, N_NODES, N_NODES, 64, 128,
                                                GMODE_F32, Hb, nullptr, nullptr, nullptr, nullptr);
    k_es_ed<<<blk((long long)N_NODES*2,T),T,0,stream>>>(Hb, g2as, g2ad, ES, ED, 2, 64);
    hipMemsetAsync(MX, 0xFF, (size_t)N_NODES*2*4, stream);
    hipMemsetAsync(DEN, 0,   (size_t)N_NODES*2*4, stream);
    hipMemsetAsync(ACC, 0,   (size_t)N_NODES*128*4, stream);
    k_edge_a <<<blk(ET*2,T),T,0,stream>>>(ei, 2, ES, ED, ESCR, MX);
    k_edge_b1<<<blk(ET*2,T),T,0,stream>>>(ei, 2, ESCR, MX, DEN);
    k_edge_b2<<<blk(ET*128,T),T,0,stream>>>(ei, 2, 7, 6, ESCR, Hb, ACC);
    k_node_fin<<<blk((long long)N_NODES*128,T),T,0,stream>>>(ACC, DEN, g2b, 2, 7, 6, XH, 0, batch, POOL, CNT);

    // ---- GAT layer 3 (H=1, C=128, HC=128): h = XH[N,128] @ W3T; fused mean-pool ----
    k_gemm_wmma<<<N_NODES/16, 256, 0, stream>>>(XH, W3T, 128, N_NODES, N_NODES, 128, 128,
                                                GMODE_F32, Hb, nullptr, nullptr, nullptr, nullptr);
    k_es_ed<<<blk((long long)N_NODES,T),T,0,stream>>>(Hb, g3as, g3ad, ES, ED, 1, 128);
    hipMemsetAsync(MX, 0xFF, (size_t)N_NODES*4, stream);
    hipMemsetAsync(DEN, 0,   (size_t)N_NODES*4, stream);
    hipMemsetAsync(ACC, 0,   (size_t)N_NODES*128*4, stream);
    k_edge_a <<<blk(ET,T),T,0,stream>>>(ei, 1, ES, ED, ESCR, MX);
    k_edge_b1<<<blk(ET,T),T,0,stream>>>(ei, 1, ESCR, MX, DEN);
    k_edge_b2<<<blk(ET*128,T),T,0,stream>>>(ei, 1, 7, 7, ESCR, Hb, ACC);
    k_node_fin<<<blk((long long)N_NODES*128,T),T,0,stream>>>(ACC, DEN, g3b, 1, 7, 7, XH, 1, batch, POOL, CNT);

    // ---- drug vector: mean pool + BN -> Z[:, 0:128] ----
    k_pool_fin<<<blk(BATCH_B*128,T),T,0,stream>>>(POOL, CNT, DSC, DSH, Z);

    // ---- protein branch: embed -> 3x implicit-GEMM conv (WMMA) ----
    k_prot_embed<<<blk((long long)BATCH_B*PROT_L*32,T),T,0,stream>>>(protein, emb_prot, PE);
    // conv1: M = 512*510, K=96, O=64
    k_gemm_wmma<<<(BATCH_B*510)/16, 128, 0, stream>>>(PE, WC1T, 96, 510, 512, 32, 64,
                                                      GMODE_AFF, nullptr, Y1, SC1, SH1, nullptr);
    // conv2: M = 512*508, K=192, O=128
    k_gemm_wmma<<<(BATCH_B*508)/16, 256, 0, stream>>>(Y1, WC2T, 192, 508, 510, 64, 128,
                                                      GMODE_AFF, nullptr, Y2, SC2, SH2, nullptr);
    // conv3 + global max pool: M = 512*506, K=384, O=128
    k_gemm_wmma<<<(BATCH_B*506)/16, 256, 0, stream>>>(Y2, WC3T, 384, 506, 508, 128, 128,
                                                      GMODE_MAX, nullptr, nullptr, SC3, SH3, PROT);
    k_prot_to_z<<<blk(BATCH_B*128,T),T,0,stream>>>(PROT, Z);

    // ---- head MLP (WMMA) ----
    k_gemm_wmma<<<BATCH_B/16, 256, 0, stream>>>(Z, F1T, 256, BATCH_B, BATCH_B, 256, 128,
                                                GMODE_AFF, nullptr, Z1, SCF1, SHF1, nullptr);
    k_gemm_wmma<<<BATCH_B/16, 128, 0, stream>>>(Z1, F2T, 128, BATCH_B, BATCH_B, 128, 64,
                                                GMODE_AFF, nullptr, Z2, SCF2, SHF2, nullptr);
    k_out_head<<<blk(BATCH_B,T),T,0,stream>>>(Z2, outW, outb, (float*)d_out);
}
